// Model_49761491091552
// MI455X (gfx1250) — matrix operand order repaired and checked by reading, not compiled, not run on hardware
//
#include <hip/hip_runtime.h>
#include <math.h>

// ---------------------------------------------------------------------------
// 2s-AGCN unit_gcn forward for MI455X (gfx1250), bf16 WMMA + fp32 accumulate.
// ---------------------------------------------------------------------------

typedef __bf16 bf16_t;
typedef bf16_t v16bf __attribute__((ext_vector_type(16)));
typedef float  v8f   __attribute__((ext_vector_type(8)));

#define WMMA_BF16(a, b, c) \
  __builtin_amdgcn_wmma_f32_16x16x32_bf16(false, (a), false, (b), (short)0, (c), false, false)

constexpr int N_  = 32;
constexpr int C_  = 64;
constexpr int T_  = 300;
constexpr int V_  = 25;
constexpr int S_  = 3;
constexpr int IN_ = 16;          // inter channels
constexpr float DINV = 1.0f / (float)(IN_ * T_);   // 1/4800

// gfx1250 WMMA 16x16x32 bf16 VGPR layout maps (per ISA 7.12.2):
// A (16x32): M = lane&15 ; K(slot) = slot + 8*(slot>=8) + 8*(lane>=16)
// B (32x16): N = lane&15 ; K(slot) = slot + 8*(slot>=8) + 8*(lane>=16)
// C/D f32  : N = lane&15 ; M(vgpr r) = r + 8*(lane>=16)
__device__ __forceinline__ int akmap(int lane, int slot) {
  return slot + ((slot >= 8) ? 8 : 0) + ((lane & 16) ? 8 : 0);
}
__device__ __forceinline__ int bkmap(int lane, int slot) {
  return slot + ((slot >= 8) ? 8 : 0) + ((lane & 16) ? 8 : 0);
}

// ===========================================================================
// Kernel 1: per (n,s) compute Ai = softmax_v( (Wa x)^T (Wb x) / d ) + A + PA
// Output: AiWs[n][s][32][32] fp32, zero-padded outside 25x25 (K-padding!).
// ===========================================================================
struct K1SM {
  bf16_t xb[C_][256];            // x chunk, [c][tl*25+v], cols 250..255 zero
  bf16_t wa[IN_][C_];
  bf16_t wb[IN_][C_];
  float  bav[IN_], bbv[IN_];
  union {
    struct { bf16_t a[IN_][256]; bf16_t b[IN_][256]; } ab;     // 16 KB
    struct { float spart[8][16][16]; float sfull[32][32]; } fin; // 12 KB
  } u;
};

__global__ __launch_bounds__(256) void k1_attention(
    const float* __restrict__ x,  const float* __restrict__ A,
    const float* __restrict__ PA, const float* __restrict__ Wa,
    const float* __restrict__ ba, const float* __restrict__ Wb,
    const float* __restrict__ bb, float* __restrict__ AiWs)
{
  __shared__ K1SM sm;
  const int n    = blockIdx.x / S_;
  const int s    = blockIdx.x % S_;
  const int tid  = threadIdx.x;
  const int lane = tid & 31;
  const int wid  = tid >> 5;
  const int hi   = (lane & 16) ? 1 : 0;

  // stage projection weights / biases (bf16)
  for (int e = tid; e < IN_ * C_; e += 256) {
    sm.wa[0][e] = (bf16_t)Wa[s * IN_ * C_ + e];
    sm.wb[0][e] = (bf16_t)Wb[s * IN_ * C_ + e];
  }
  if (tid < IN_) { sm.bav[tid] = ba[s * IN_ + tid]; sm.bbv[tid] = bb[s * IN_ + tid]; }

  // score tile ownership: tiles (2x2 of 16x16), 2 waves per tile split over k
  const int tIdx = wid & 3;
  const int mt   = tIdx >> 1;
  const int nt   = tIdx & 1;
  const int par  = wid >> 2;
  v8f acc = {0.f, 0.f, 0.f, 0.f, 0.f, 0.f, 0.f, 0.f};

  const float* xn = x + (size_t)n * C_ * T_ * V_;

  for (int ch = 0; ch < 30; ++ch) {        // T chunks of 10
    const int t0 = ch * 10;
    __syncthreads();                       // previous readers of xb / a/b done
    for (int e = tid; e < C_ * 256; e += 256) {
      const int c = e >> 8, col = e & 255;
      float v = 0.f;
      if (col < 250) v = xn[(c * T_ + t0 + col / 25) * V_ + (col % 25)];
      sm.xb[c][col] = (bf16_t)v;
    }
    __syncthreads();

    // projections a,b: M=16(i) x K=64(c) x N=256(tv). 32 n-tile jobs over 8 waves.
    for (int j = wid * 4; j < wid * 4 + 4; ++j) {
      const bool isA  = (j < 16);
      const int ntile = isA ? j : (j - 16);
      v8f d = {0.f, 0.f, 0.f, 0.f, 0.f, 0.f, 0.f, 0.f};
      for (int ks = 0; ks < 2; ++ks) {
        v16bf av, bv;
#pragma unroll
        for (int sl = 0; sl < 16; ++sl) {
          const int ka = akmap(lane, sl) + 32 * ks;
          av[sl] = isA ? sm.wa[lane & 15][ka] : sm.wb[lane & 15][ka];
          const int kb = bkmap(lane, sl) + 32 * ks;
          bv[sl] = sm.xb[kb][(lane & 15) + 16 * ntile];
        }
        d = WMMA_BF16(av, bv, d);
      }
#pragma unroll
      for (int r = 0; r < 8; ++r) {
        const int m = r + 8 * hi;
        const float val = d[r] + (isA ? sm.bav[m] : sm.bbv[m]);
        (isA ? sm.u.ab.a : sm.u.ab.b)[m][16 * ntile + (lane & 15)] = (bf16_t)val;
      }
    }
    __syncthreads();

    // scores += a^T b over chunk contraction k' = i*10 + tl (160 -> 5 k-tiles)
    for (int kk = par; kk < 5; kk += 2) {
      v16bf av, bv;
#pragma unroll
      for (int sl = 0; sl < 16; ++sl) {
        const int ka = akmap(lane, sl) + 32 * kk;
        const int ia = ka / 10, tla = ka % 10;
        const int vr = (lane & 15) + 16 * mt;
        av[sl] = (vr < 25) ? sm.u.ab.a[ia][tla * 25 + vr] : (bf16_t)0.0f;
        const int kb = bkmap(lane, sl) + 32 * kk;
        const int ib = kb / 10, tlb = kb % 10;
        const int wc = (lane & 15) + 16 * nt;
        bv[sl] = (wc < 25) ? sm.u.ab.b[ib][tlb * 25 + wc] : (bf16_t)0.0f;
      }
      acc = WMMA_BF16(av, bv, acc);
    }
  }
  __syncthreads();

  // reduce the two k-halves per tile, scale by 1/d
#pragma unroll
  for (int r = 0; r < 8; ++r)
    sm.u.fin.spart[wid][r + 8 * hi][lane & 15] = acc[r];
  __syncthreads();
  for (int e = tid; e < 1024; e += 256) {
    const int m = e >> 5, nn = e & 31;
    const int tI = (m >> 4) * 2 + (nn >> 4);
    const float v = sm.u.fin.spart[tI][m & 15][nn & 15] +
                    sm.u.fin.spart[tI + 4][m & 15][nn & 15];
    sm.u.fin.sfull[m][nn] = v * DINV;
  }
  __syncthreads();

  // softmax over v (axis -2) per column w, then + A + PA; zero-pad to 32x32
  if (tid < 32) {
    const int w = tid;
    float mx = -1e30f, sum = 0.f;
    if (w < 25) {
      for (int v = 0; v < 25; ++v) mx = fmaxf(mx, sm.u.fin.sfull[v][w]);
      for (int v = 0; v < 25; ++v) sum += __expf(sm.u.fin.sfull[v][w] - mx);
    }
    const float inv = (w < 25) ? (1.f / sum) : 0.f;
    float* dst = AiWs + (size_t)(n * S_ + s) * 32 * 32;
    for (int v = 0; v < 32; ++v) {
      float val = 0.f;
      if (v < 25 && w < 25) {
        const float att = __expf(sm.u.fin.sfull[v][w] - mx) * inv;
        val = att + A[(s * V_ + v) * V_ + w] + PA[(s * V_ + v) * V_ + w];
      }
      dst[v * 32 + w] = val;
    }
  }
}

// ===========================================================================
// Kernel 2: fused  z = Wd[s] @ x ;  y += z @ Ai[s]  (sum over s)
//           out = relu( y*bn_scale + (beta + bn_scale*sum_s bd) + x )
// One block per (n, 5-wide T chunk).
// ===========================================================================
constexpr int TC2 = 5;           // t per chunk; N-dim = 125 -> pad 128

struct K2SM {
  bf16_t xb[C_][128];            // [c][tl*25+v], cols 125..127 zero
  bf16_t z[C_][128];             // z for current s
  bf16_t wd[S_][C_][C_];         // all Wd in bf16
  bf16_t ai[S_][32][32];         // Ai bf16, zero padded (K padding exact)
  float  scalec[C_];
  float  biasc[C_];
};

__global__ __launch_bounds__(256) void k2_fused(
    const float* __restrict__ x,  const float* __restrict__ Wd,
    const float* __restrict__ bd, const float* __restrict__ gamma,
    const float* __restrict__ beta, const float* __restrict__ AiWs,
    float* __restrict__ out)
{
  __shared__ K2SM sm;
  const int n    = blockIdx.y;
  const int t0   = blockIdx.x * TC2;
  const int tid  = threadIdx.x;
  const int lane = tid & 31;
  const int wid  = tid >> 5;
  const int hi   = (lane & 16) ? 1 : 0;

  for (int e = tid; e < S_ * C_ * C_; e += 256) sm.wd[0][0][e] = (bf16_t)Wd[e];
  const float* ain = AiWs + (size_t)n * S_ * 1024;
  for (int e = tid; e < S_ * 1024; e += 256) sm.ai[0][0][e] = (bf16_t)ain[e];

  const float* xn = x + (size_t)n * C_ * T_ * V_;
  for (int e = tid; e < C_ * 128; e += 256) {
    const int c = e >> 7, col = e & 127;
    float v = 0.f;
    if (col < TC2 * V_) v = xn[(c * T_ + t0 + col / 25) * V_ + (col % 25)];
    sm.xb[c][col] = (bf16_t)v;
  }
  if (tid < C_) {
    const float sc = gamma[tid] * rsqrtf(1.0f + 1e-5f);
    sm.scalec[tid] = sc;
    sm.biasc[tid]  = beta[tid] + (bd[tid] + bd[C_ + tid] + bd[2 * C_ + tid]) * sc;
  }
  __syncthreads();

  // persistent y accumulators: 40 tiles (20 m-tiles x 2 n-tiles), 5 per wave
  v8f yacc[5];
#pragma unroll
  for (int j = 0; j < 5; ++j) yacc[j] = (v8f){0.f, 0.f, 0.f, 0.f, 0.f, 0.f, 0.f, 0.f};

  for (int s = 0; s < S_; ++s) {
    // step A: z = Wd[s] @ x   (M=64, K=64, N=128) -> 32 tiles, 4 per wave
    for (int jj = 0; jj < 4; ++jj) {
      const int tile = wid * 4 + jj;
      const int mtA = tile >> 3, ntA = tile & 7;
      v8f d = {0.f, 0.f, 0.f, 0.f, 0.f, 0.f, 0.f, 0.f};
      for (int ks = 0; ks < 2; ++ks) {
        v16bf av, bv;
#pragma unroll
        for (int sl = 0; sl < 16; ++sl) {
          const int ka = akmap(lane, sl) + 32 * ks;
          av[sl] = sm.wd[s][(lane & 15) + 16 * mtA][ka];
          const int kb = bkmap(lane, sl) + 32 * ks;
          bv[sl] = sm.xb[kb][(lane & 15) + 16 * ntA];
        }
        d = WMMA_BF16(av, bv, d);
      }
#pragma unroll
      for (int r = 0; r < 8; ++r)
        sm.z[mtA * 16 + r + 8 * hi][ntA * 16 + (lane & 15)] = (bf16_t)d[r];
    }
    __syncthreads();

    // step B: y += z @ Ai[s]  (M=320 rows (c,tl), K=32(v pad), N=32(w pad))
    for (int jj = 0; jj < 5; ++jj) {
      const int g = wid * 5 + jj;            // 0..39
      const int mtB = g >> 1, nt2 = g & 1;
      v16bf av, bv;
#pragma unroll
      for (int sl = 0; sl < 16; ++sl) {
        const int v = akmap(lane, sl);       // k index = graph node v
        const int m = mtB * 16 + (lane & 15);
        const int c = m / 5, tl = m % 5;
        av[sl] = (v < 25) ? sm.z[c][tl * 25 + v] : (bf16_t)0.0f;
        const int kb = bkmap(lane, sl);
        bv[sl] = sm.ai[s][kb][(lane & 15) + 16 * nt2];
      }
      yacc[jj] = WMMA_BF16(av, bv, yacc[jj]);
    }
    __syncthreads();
  }

  // epilogue: BN(eval) + folded bd + residual + ReLU
  float* outn = out + (size_t)n * C_ * T_ * V_;
  for (int jj = 0; jj < 5; ++jj) {
    const int g = wid * 5 + jj;
    const int mtB = g >> 1, nt2 = g & 1;
    const int w = nt2 * 16 + (lane & 15);
    if (w < 25) {
#pragma unroll
      for (int r = 0; r < 8; ++r) {
        const int m = mtB * 16 + r + 8 * hi;
        const int c = m / 5, tl = m % 5;
        const size_t idx = ((size_t)c * T_ + t0 + tl) * V_ + w;
        const float val = yacc[jj][r] * sm.scalec[c] + sm.biasc[c] + xn[idx];
        outn[idx] = fmaxf(val, 0.f);
      }
    }
  }
}

// ===========================================================================
extern "C" void kernel_launch(void* const* d_in, const int* in_sizes, int n_in,
                              void* d_out, int out_size, void* d_ws, size_t ws_size,
                              hipStream_t stream)
{
  const float* x     = (const float*)d_in[0];
  const float* A     = (const float*)d_in[1];
  const float* PA    = (const float*)d_in[2];
  const float* Wa    = (const float*)d_in[3];
  const float* ba    = (const float*)d_in[4];
  const float* Wb    = (const float*)d_in[5];
  const float* bb    = (const float*)d_in[6];
  const float* Wd    = (const float*)d_in[7];
  const float* bd    = (const float*)d_in[8];
  const float* gamma = (const float*)d_in[9];
  const float* beta  = (const float*)d_in[10];
  float* out  = (float*)d_out;
  float* AiWs = (float*)d_ws;           // needs N*S*32*32*4 = 393216 bytes

  (void)in_sizes; (void)n_in; (void)out_size; (void)ws_size;

  hipLaunchKernelGGL(k1_attention, dim3(N_ * S_), dim3(256), 0, stream,
                     x, A, PA, Wa, ba, Wb, bb, AiWs);
  hipLaunchKernelGGL(k2_fused, dim3(T_ / TC2, N_), dim3(256), 0, stream,
                     x, Wd, bd, gamma, beta, AiWs, out);
}
